// Recurrent_ALIFLayer2D_77515569758390
// MI455X (gfx1250) — matrix operand order repaired and checked by reading, not compiled, not run on hardware
//
#include <hip/hip_runtime.h>

typedef _Float16 h8   __attribute__((ext_vector_type(8)));
typedef _Float16 v16h __attribute__((ext_vector_type(16)));
typedef float    v8f  __attribute__((ext_vector_type(8)));

#define FDIM   256
#define TSTEPS 64
#define HWDIM  256      // H*W = 16*16
#define ROWS_PER_BLOCK 16
#define NTHREADS 256

union V16U { v16h v; h8 h[2]; };

// ---------------- prep: w_rec (f32, [k][n] row-major) -> f16 column-major [n][k] ----------------
__global__ __launch_bounds__(NTHREADS)
void alif_prep_w(const float* __restrict__ w_rec, _Float16* __restrict__ w_cm) {
    int idx = blockIdx.x * NTHREADS + threadIdx.x;   // 65536 total
    int n = idx >> 8;
    int k = idx & 255;
    w_cm[n * FDIM + k] = (_Float16)w_rec[k * FDIM + n];
}

// ---------------- main scan: each block owns 16 rows for all 64 timesteps ----------------
__global__ __launch_bounds__(NTHREADS, 1)
void alif_scan(const float* __restrict__ x,
               const _Float16* __restrict__ w_cm,
               const float* __restrict__ p_d,
               const float* __restrict__ p_adp,
               const float* __restrict__ p_beta,
               float* __restrict__ out) {
    // spikes from previous step, row-major [16][256] f16 (8 KB)
    __shared__ _Float16 Sbuf[ROWS_PER_BLOCK * FDIM];

    const float dk   = *p_d;
    const float adp  = *p_adp;
    const float beta = *p_beta;

    const int tid  = threadIdx.x;
    const int lane = tid & 31;
    const int wave = tid >> 5;
    const int lo16 = lane & 15;
    const int hi   = lane >> 4;        // 0 or 1 (half-wave)
    const int n0w  = wave * 32;        // this wave's 32 output columns

    // zero the spike buffer (initial carry s = 0)
    for (int i = tid; i < ROWS_PER_BLOCK * FDIM / 2; i += NTHREADS)
        ((unsigned int*)Sbuf)[i] = 0u;

    // ---- load this wave's W slice as WMMA B fragments, register-resident across the scan ----
    // B (32x16 f16): lane -> col n = lo16; K contiguous block = k0*32 + hi*16 .. +15
    v16h Breg[2][8];
#pragma unroll
    for (int tl = 0; tl < 2; ++tl) {
        const _Float16* colp = w_cm + (size_t)(n0w + 16 * tl + lo16) * FDIM + hi * 16;
#pragma unroll
        for (int k0 = 0; k0 < 8; ++k0)
            Breg[tl][k0] = *(const v16h*)(colp + 32 * k0);   // 32B contiguous, aligned
    }

    // C/D layout: tile tl, vgpr j -> row m = j + 8*hi, col = n0w + 16*tl + lo16
    const int blk = blockIdx.x;
    const int bb  = blk >> 4;                 // batch index (16 blocks per image)
    int rowoff[8];
#pragma unroll
    for (int j = 0; j < 8; ++j) {
        int m  = j + 8 * hi;
        int hw = ((blk & 15) << 4) + m;
        rowoff[j] = bb * (TSTEPS * HWDIM * FDIM) + hw * FDIM;
    }
    const int col0 = n0w + lo16;
    const int col1 = n0w + 16 + lo16;

    float v[2][8], th[2][8];
#pragma unroll
    for (int tl = 0; tl < 2; ++tl)
#pragma unroll
        for (int j = 0; j < 8; ++j) { v[tl][j] = 0.0f; th[tl][j] = 0.0f; }

    __syncthreads();

    // A fragment addressing (16x32 f16): lane -> row m_a = lo16,
// K blocks: [k0*32 + hi*16 .. +7] and [k0*32 + hi*16 + 8 .. +15]
    const int m_a  = lo16;
    const int kb_a = hi * 16;

    for (int t = 0; t < TSTEPS; ++t) {
        // ---- recurrent matmul: rec = S_prev(16x256) @ W, this wave's 32 cols ----
        v8f acc0 = {0.f, 0.f, 0.f, 0.f, 0.f, 0.f, 0.f, 0.f};
        v8f acc1 = {0.f, 0.f, 0.f, 0.f, 0.f, 0.f, 0.f, 0.f};
#pragma unroll
        for (int k0 = 0; k0 < 8; ++k0) {
            const _Float16* ap = Sbuf + m_a * FDIM + 32 * k0 + kb_a;
            V16U a;
            a.h[0] = *(const h8*)(ap);        // K = k0*32 + hi*16 .. +7
            a.h[1] = *(const h8*)(ap + 8);    // K = k0*32 + hi*16 + 8 .. +15
            acc0 = __builtin_amdgcn_wmma_f32_16x16x32_f16(
                       false, a.v, false, Breg[0][k0], (short)0, acc0, false, false);
            acc1 = __builtin_amdgcn_wmma_f32_16x16x32_f16(
                       false, a.v, false, Breg[1][k0], (short)0, acc1, false, false);
        }

        __syncthreads();   // everyone done reading S_prev before we overwrite it

        // ---- elementwise ALIF update (theta carried pre-updated for next step) ----
        const int toff = t * (HWDIM * FDIM);
#pragma unroll
        for (int tl = 0; tl < 2; ++tl) {
            v8f acc = tl ? acc1 : acc0;
            const int col = tl ? col1 : col0;
#pragma unroll
            for (int j = 0; j < 8; ++j) {
                const int idx = rowoff[j] + toff + col;
                const float vth = 0.5f + th[tl][j];
                float vn = v[tl][j] * dk + x[idx] + acc[j];
                const float s = (vn - vth) > 0.0f ? 1.0f : 0.0f;
                vn -= s * vth;
                v[tl][j]  = vn;
                th[tl][j] = th[tl][j] * adp + s * beta;   // = theta used at step t+1
                out[idx] = s;
                const int m = j + 8 * hi;
                Sbuf[m * FDIM + col] = (_Float16)s;
            }
        }
        __syncthreads();   // new spikes visible to all waves before next matmul
    }
}

extern "C" void kernel_launch(void* const* d_in, const int* in_sizes, int n_in,
                              void* d_out, int out_size, void* d_ws, size_t ws_size,
                              hipStream_t stream) {
    const float* x      = (const float*)d_in[0];
    const float* w_rec  = (const float*)d_in[1];
    const float* p_d    = (const float*)d_in[2];
    const float* p_adp  = (const float*)d_in[3];
    const float* p_beta = (const float*)d_in[4];
    // d_in[5] = hp_alpha: only used by the surrogate gradient (backward), not forward.

    _Float16* w_cm = (_Float16*)d_ws;   // 256*256 f16 = 128 KB scratch

    alif_prep_w<<<256, NTHREADS, 0, stream>>>(w_rec, w_cm);
    alif_scan<<<(2048 / ROWS_PER_BLOCK), NTHREADS, 0, stream>>>(
        x, w_cm, p_d, p_adp, p_beta, (float*)d_out);
}
